// DualLstm_48146583388462
// MI455X (gfx1250) — matrix operand order repaired and checked by reading, not compiled, not run on hardware
//
#include <hip/hip_runtime.h>
#include <hip/hip_bf16.h>

// MI455X (gfx1250) implementation of the dual-LSTM captioning decoder.
//
// Roofline: ~176 GFLOP total vs ~528 MB of weights re-streamed for each of the
// 20 sequential timesteps => bandwidth bound (16 flop/byte in fp32, B=32).
// Strategy: convert all weights to bf16 WMMA B-fragments once per launch,
// then run every GEMM with v_wmma_f32_16x16x32_bf16 (fp32 accum), halving the
// steady-state HBM traffic. The attention MLP (Wv/Wh/Wa) is skipped entirely:
// the reference applies softmax over a singleton axis, so attention weights
// are identically 1 and v_hat = sum(features, axis=1) = 36 * v_bar (t-invariant).

#define D_   2048
#define V_   10000
#define B_   32
#define X_   36
#define T_   20

typedef __attribute__((ext_vector_type(16))) __bf16 v16bf;
typedef __attribute__((ext_vector_type(8)))  float  v8f;

union Frag16 { float4 f4[2]; v16bf v; };

// ---------------------------------------------------------------------------
// Weight conversion: fp32 row-major [N, Ktot] (split across W1 [N,K1] and
// W2 [N,Ktot-K1]) -> bf16 WMMA B-fragments.
// Fragment (n_tile, k_tile) = 512 bf16 (1KB), lane-major:
//   lane L<16  : W[n_tile*16+L,      k_tile*32 + 0..15]
//   lane L>=16 : W[n_tile*16+L-16,   k_tile*32 + 16..31]
// Fragment order: frag = n_tile * KT + k_tile  (k-contiguous streaming).
// ---------------------------------------------------------------------------
__global__ void conv_weights(const float* __restrict__ W1, int K1,
                             const float* __restrict__ W2, int Ktot,
                             __bf16* __restrict__ dst, int nfrags) {
  int gtid = blockIdx.x * 256 + threadIdx.x;
  int frag = gtid >> 5;
  if (frag >= nfrags) return;
  int lane  = gtid & 31;
  int KT    = Ktot >> 5;
  int ntile = frag / KT;
  int ktile = frag - ntile * KT;
  int row   = ntile * 16 + (lane & 15);
  int kc    = ktile * 32 + ((lane >> 4) << 4);   // 0 or +16
  const float* src;
  if (kc < K1) src = W1 + (size_t)row * K1 + kc;
  else         src = W2 + (size_t)row * (Ktot - K1) + (kc - K1);
  __bf16 tmp[16];
#pragma unroll
  for (int i = 0; i < 16; ++i) tmp[i] = (__bf16)src[i];
  float4* d = (float4*)(dst + (size_t)frag * 512 + lane * 16);
  d[0] = *(float4*)&tmp[0];
  d[1] = *(float4*)&tmp[8];
}

// v_bar = mean(features, axis=1) -> bf16 into xa[:,2048:4096]
// v_hat = sum(features, axis=1)  -> bf16 into xl[:,2048:4096]
__global__ void vbar_kernel(const float* __restrict__ feat,
                            __bf16* __restrict__ xa, __bf16* __restrict__ xl) {
  int b = blockIdx.x;
  for (int k = threadIdx.x; k < D_; k += 256) {
    float s = 0.f;
#pragma unroll 4
    for (int x = 0; x < X_; ++x)
      s += feat[((size_t)b * X_ + x) * D_ + k];
    xa[(size_t)b * 8192 + 2048 + k] = (__bf16)(s * (1.0f / 36.0f));
    xl[(size_t)b * 6144 + 2048 + k] = (__bf16)s;
  }
}

// Zero fp32 cell states and bf16 recurrent-h regions (65536 threads).
__global__ void init_zero(float* __restrict__ ca, float* __restrict__ cl,
                          __bf16* __restrict__ xa, __bf16* __restrict__ xl) {
  int t = blockIdx.x * 256 + threadIdx.x;          // 0 .. 65535
  ca[t] = 0.f;
  cl[t] = 0.f;
  int b = t >> 11, k = t & 2047;
  xa[(size_t)b * 8192 + k]        = (__bf16)0.f;   // h_l slot
  xa[(size_t)b * 8192 + 6144 + k] = (__bf16)0.f;   // h_a slot
  xl[(size_t)b * 6144 + k]        = (__bf16)0.f;   // h_a slot
  xl[(size_t)b * 6144 + 4096 + k] = (__bf16)0.f;   // h_l slot
}

__global__ void bias_comb(const float* __restrict__ biha, const float* __restrict__ bhha,
                          const float* __restrict__ bihl, const float* __restrict__ bhhl,
                          float* __restrict__ bca, float* __restrict__ bcl) {
  int n = blockIdx.x * 256 + threadIdx.x;
  if (n < 8192) { bca[n] = biha[n] + bhha[n]; bcl[n] = bihl[n] + bhhl[n]; }
}

// e_t = emb[caption[:, t]] -> bf16 into xa[:,4096:6144]
__global__ void emb_gather(const float* __restrict__ emb, const int* __restrict__ caption,
                           __bf16* __restrict__ xa, int t) {
  int b = blockIdx.x;
  int cap = caption[b * T_ + t];
  const float* src = emb + (size_t)cap * D_;
  for (int k = threadIdx.x; k < D_; k += 256)
    xa[(size_t)b * 8192 + 4096 + k] = (__bf16)src[k];
}

// ---------------------------------------------------------------------------
// Fused LSTM cell: z = x_bf16 @ Wcomb^T + bcomb ; gates ; state update.
// grid.x = 128 (16-column tiles of the 2048 hidden units), block = 128 (4 waves).
// Wave w computes gate block w (i,f,g,o) for the same 16 hidden columns:
//   n_tile = w*128 + blockIdx.x.  Row-stride of xbf equals K.
// Epilogue writes c (fp32, in place) and h as bf16 into two activation slots.
// ---------------------------------------------------------------------------
__global__ void lstm_gemm_gates(const __bf16* __restrict__ xbf, int K,
                                const __bf16* __restrict__ wfrag,
                                const float* __restrict__ bcomb,
                                float* __restrict__ c_state,
                                __bf16* __restrict__ hdst1, int stride1, int off1,
                                __bf16* __restrict__ hdst2, int stride2, int off2) {
  int wv   = threadIdx.x >> 5;
  int lane = threadIdx.x & 31;
  int n_tile = wv * 128 + blockIdx.x;
  int KT = K >> 5;

  const __bf16* bp = wfrag + (size_t)n_tile * KT * 512 + lane * 16;
  int arow   = lane & 15;
  int kshift = (lane < 16) ? 0 : 16;
  const __bf16* a0base = xbf + (size_t)arow * K + kshift;
  const __bf16* a1base = a0base + (size_t)16 * K;

  v8f c0 = {}; v8f c1 = {};
  for (int kt = 0; kt < KT; ++kt) {
    Frag16 bf, a0, a1;
    const float4* pb = (const float4*)bp;
    bf.f4[0] = pb[0]; bf.f4[1] = pb[1];
    const float4* p0 = (const float4*)(a0base + (size_t)kt * 32);
    a0.f4[0] = p0[0]; a0.f4[1] = p0[1];
    const float4* p1 = (const float4*)(a1base + (size_t)kt * 32);
    a1.f4[0] = p1[0]; a1.f4[1] = p1[1];
    c0 = __builtin_amdgcn_wmma_f32_16x16x32_bf16(false, a0.v, false, bf.v,
                                                 (short)0, c0, false, false);
    c1 = __builtin_amdgcn_wmma_f32_16x16x32_bf16(false, a1.v, false, bf.v,
                                                 (short)0, c1, false, false);
    bp += 512;
  }

  float bias = bcomb[n_tile * 16 + (lane & 15)];
  __shared__ float zsh[4][32][16];
  int col = lane & 15;
  int mb  = (lane < 16) ? 0 : 8;
#pragma unroll
  for (int r = 0; r < 8; ++r) {
    zsh[wv][mb + r][col]      = c0[r] + bias;
    zsh[wv][16 + mb + r][col] = c1[r] + bias;
  }
  __syncthreads();

#pragma unroll
  for (int q = 0; q < 4; ++q) {
    int e  = threadIdx.x + q * 128;          // 0..511 = (batch, col)
    int b  = e >> 4;
    int cc = e & 15;
    int j  = blockIdx.x * 16 + cc;           // hidden unit index
    float zi = zsh[0][b][cc], zf = zsh[1][b][cc];
    float zg = zsh[2][b][cc], zo = zsh[3][b][cc];
    float cold = c_state[b * D_ + j];
    float ig = 1.f / (1.f + expf(-zi));
    float fg = 1.f / (1.f + expf(-zf));
    float gg = tanhf(zg);
    float og = 1.f / (1.f + expf(-zo));
    float cn = fg * cold + ig * gg;
    float hn = og * tanhf(cn);
    c_state[b * D_ + j] = cn;
    __bf16 hb = (__bf16)hn;
    hdst1[(size_t)b * stride1 + off1 + j] = hb;
    hdst2[(size_t)b * stride2 + off2 + j] = hb;
  }
}

// logits = h_l @ Wfc^T + bfc ; grid = 625 n-tiles, block = 32 (1 wave).
__global__ void logits_gemm(const __bf16* __restrict__ hbf, int rowstride,
                            const __bf16* __restrict__ wfrag,
                            const float* __restrict__ bfc,
                            float* __restrict__ logits) {
  int lane   = threadIdx.x & 31;
  int n_tile = blockIdx.x;
  const int KT = D_ / 32;   // 64
  const __bf16* bp = wfrag + (size_t)n_tile * KT * 512 + lane * 16;
  int arow   = lane & 15;
  int kshift = (lane < 16) ? 0 : 16;
  const __bf16* a0base = hbf + (size_t)arow * rowstride + kshift;
  const __bf16* a1base = a0base + (size_t)16 * rowstride;

  v8f c0 = {}; v8f c1 = {};
  for (int kt = 0; kt < KT; ++kt) {
    Frag16 bf, a0, a1;
    const float4* pb = (const float4*)bp;
    bf.f4[0] = pb[0]; bf.f4[1] = pb[1];
    const float4* p0 = (const float4*)(a0base + (size_t)kt * 32);
    a0.f4[0] = p0[0]; a0.f4[1] = p0[1];
    const float4* p1 = (const float4*)(a1base + (size_t)kt * 32);
    a1.f4[0] = p1[0]; a1.f4[1] = p1[1];
    c0 = __builtin_amdgcn_wmma_f32_16x16x32_bf16(false, a0.v, false, bf.v,
                                                 (short)0, c0, false, false);
    c1 = __builtin_amdgcn_wmma_f32_16x16x32_bf16(false, a1.v, false, bf.v,
                                                 (short)0, c1, false, false);
    bp += 512;
  }
  int col = lane & 15;
  int mb  = (lane < 16) ? 0 : 8;
  float bias = bfc[n_tile * 16 + col];
#pragma unroll
  for (int r = 0; r < 8; ++r) {
    logits[(size_t)(mb + r) * V_ + n_tile * 16 + col]      = c0[r] + bias;
    logits[(size_t)(16 + mb + r) * V_ + n_tile * 16 + col] = c1[r] + bias;
  }
}

// log_softmax over V per batch row; writes d_out[b, t, :].
__global__ void logsoftmax_kernel(const float* __restrict__ logits,
                                  float* __restrict__ out, int t) {
  int b = blockIdx.x;
  const float* row = logits + (size_t)b * V_;
  __shared__ float red[256];
  float m = -1e30f;
  for (int v = threadIdx.x; v < V_; v += 256) m = fmaxf(m, row[v]);
  red[threadIdx.x] = m; __syncthreads();
  for (int s = 128; s > 0; s >>= 1) {
    if (threadIdx.x < s) red[threadIdx.x] = fmaxf(red[threadIdx.x], red[threadIdx.x + s]);
    __syncthreads();
  }
  m = red[0]; __syncthreads();
  float s = 0.f;
  for (int v = threadIdx.x; v < V_; v += 256) s += expf(row[v] - m);
  red[threadIdx.x] = s; __syncthreads();
  for (int st = 128; st > 0; st >>= 1) {
    if (threadIdx.x < st) red[threadIdx.x] += red[threadIdx.x + st];
    __syncthreads();
  }
  float L = m + logf(red[0]);
  float* orow = out + ((size_t)b * T_ + t) * V_;
  for (int v = threadIdx.x; v < V_; v += 256) orow[v] = row[v] - L;
}

// ---------------------------------------------------------------------------
extern "C" void kernel_launch(void* const* d_in, const int* in_sizes, int n_in,
                              void* d_out, int out_size, void* d_ws, size_t ws_size,
                              hipStream_t stream) {
  (void)in_sizes; (void)n_in; (void)out_size; (void)ws_size;

  const float* features = (const float*)d_in[0];
  const int*   caption  = (const int*)  d_in[1];
  const float* emb      = (const float*)d_in[2];
  const float* Wih_a    = (const float*)d_in[3];
  const float* Whh_a    = (const float*)d_in[4];
  const float* bih_a    = (const float*)d_in[5];
  const float* bhh_a    = (const float*)d_in[6];
  const float* Wih_l    = (const float*)d_in[7];
  const float* Whh_l    = (const float*)d_in[8];
  const float* bih_l    = (const float*)d_in[9];
  const float* bhh_l    = (const float*)d_in[10];
  // d_in[11..16] = Wv,bv,Wh,bh,Wa,ba : provably dead (softmax over singleton).
  const float* Wfc      = (const float*)d_in[17];
  const float* bfc      = (const float*)d_in[18];
  float* out = (float*)d_out;

  // Workspace layout (all 256B-aligned offsets), total ~266 MB.
  char* ws = (char*)d_ws;
  const size_t OFF_WA  = 0;                          // 8192x8192 bf16 = 128 MB
  const size_t OFF_WL  = OFF_WA  + (size_t)134217728;// 8192x6144 bf16 =  96 MB
  const size_t OFF_WFC = OFF_WL  + (size_t)100663296;// 10000x2048 bf16 = 39 MB
  const size_t OFF_XA  = OFF_WFC + (size_t)40960000; // [32,8192] bf16
  const size_t OFF_XL  = OFF_XA  + (size_t)524288;   // [32,6144] bf16
  const size_t OFF_CA  = OFF_XL  + (size_t)393216;   // [32,2048] f32
  const size_t OFF_CL  = OFF_CA  + (size_t)262144;   // [32,2048] f32
  const size_t OFF_BA  = OFF_CL  + (size_t)262144;   // [8192] f32
  const size_t OFF_BL  = OFF_BA  + (size_t)32768;    // [8192] f32
  const size_t OFF_LG  = OFF_BL  + (size_t)32768;    // [32,10000] f32

  __bf16* wa_frag  = (__bf16*)(ws + OFF_WA);
  __bf16* wl_frag  = (__bf16*)(ws + OFF_WL);
  __bf16* wfc_frag = (__bf16*)(ws + OFF_WFC);
  __bf16* xa_bf    = (__bf16*)(ws + OFF_XA);   // [h_l | v_bar | e_t | h_a]
  __bf16* xl_bf    = (__bf16*)(ws + OFF_XL);   // [h_a | v_hat | h_l]
  float*  c_a      = (float*) (ws + OFF_CA);
  float*  c_l      = (float*) (ws + OFF_CL);
  float*  bcomb_a  = (float*) (ws + OFF_BA);
  float*  bcomb_l  = (float*) (ws + OFF_BL);
  float*  logits   = (float*) (ws + OFF_LG);

  // One-time per launch: weight conversion + state init.
  {
    int nfa = 512 * 256;   // Wih_a||Whh_a  (K=8192)
    conv_weights<<<(nfa * 32 + 255) / 256, 256, 0, stream>>>(
        Wih_a, 6144, Whh_a, 8192, wa_frag, nfa);
    int nfl = 512 * 192;   // Wih_l||Whh_l  (K=6144)
    conv_weights<<<(nfl * 32 + 255) / 256, 256, 0, stream>>>(
        Wih_l, 4096, Whh_l, 6144, wl_frag, nfl);
    int nff = 625 * 64;    // Wfc           (K=2048)
    conv_weights<<<(nff * 32 + 255) / 256, 256, 0, stream>>>(
        Wfc, 2048, (const float*)nullptr, 2048, wfc_frag, nff);

    init_zero<<<256, 256, 0, stream>>>(c_a, c_l, xa_bf, xl_bf);
    bias_comb<<<32, 256, 0, stream>>>(bih_a, bhh_a, bih_l, bhh_l, bcomb_a, bcomb_l);
    vbar_kernel<<<B_, 256, 0, stream>>>(features, xa_bf, xl_bf);
  }

  for (int t = 0; t < T_; ++t) {
    emb_gather<<<B_, 256, 0, stream>>>(emb, caption, xa_bf, t);

    // Attention LSTM: x = [h_l, v_bar, e_t, h_a] (K=8192)
    lstm_gemm_gates<<<128, 128, 0, stream>>>(
        xa_bf, 8192, wa_frag, bcomb_a, c_a,
        xl_bf, 6144, 0,       // h_a -> xl[:,0:2048]
        xa_bf, 8192, 6144);   // h_a -> xa[:,6144:8192]

    // Language LSTM: x2 = [h_a, v_hat, h_l] (K=6144)
    lstm_gemm_gates<<<128, 128, 0, stream>>>(
        xl_bf, 6144, wl_frag, bcomb_l, c_l,
        xa_bf, 8192, 0,       // h_l -> xa[:,0:2048]
        xl_bf, 6144, 4096);   // h_l -> xl[:,4096:6144]

    // Output projection from h_l (bf16 copy lives at xl[:,4096:6144]).
    logits_gemm<<<625, 32, 0, stream>>>(xl_bf + 4096, 6144, wfc_frag, bfc, logits);

    logsoftmax_kernel<<<B_, 256, 0, stream>>>(logits, out, t);
  }
}
